// Baseline_cbr_mb_38757784879352
// MI455X (gfx1250) — matrix operand order repaired and checked by reading, not compiled, not run on hardware
//
#include <hip/hip_runtime.h>

#define N_PATHS 200000
#define PATH_LEN 8
#define N_LINKS 20000
#define MAX_PL 100
#define ITERS 8

typedef __attribute__((ext_vector_type(16))) _Float16 v16h;
typedef __attribute__((ext_vector_type(8)))  _Float16 v8h;
typedef __attribute__((ext_vector_type(8)))  float    v8f;

__device__ __forceinline__ v8f wmma16(v16h a, v16h b, v8f c) {
  // D(16x16,f32) = A(16x32,f16) * B(32x16,f16) + C ; fully dense K=32: K=0..15 -> x, K=16..31 -> h
  return __builtin_amdgcn_wmma_f32_16x16x32_f16(false, a, false, b, (short)0, c, false, false);
}

// Branchless transcendental helpers: v_exp_f32 + v_rcp_f32 only (no IEEE div
// slow path, no EXEC-divergent regions inside the WMMA loops).
__device__ __forceinline__ float fast_rcp(float x) { return __builtin_amdgcn_rcpf(x); }
__device__ __forceinline__ float sigmoidf_(float x) { return fast_rcp(1.0f + __expf(-x)); }
__device__ __forceinline__ float tanhf_(float x) { return 1.0f - 2.0f * fast_rcp(1.0f + __expf(2.0f * x)); }
__device__ __forceinline__ float softplusf_(float x) { return fmaxf(x, 0.0f) + __logf(1.0f + __expf(-fabsf(x))); }

// Fused B operand (32x16 f16) for gate g: K rows 0..15 = k (input kernel),
// K rows 16..31 = rk (recurrent kernel). B layout: elem e<8 holds K=8*(lane>>4)+e,
// elem e>=8 holds K=16+8*(lane>>4)+(e-8); N = lane&15.
__device__ __forceinline__ v16h load_b_fused(const float* __restrict__ k,
                                             const float* __restrict__ rk,
                                             int g, int lane) {
  const int n = lane & 15;
  const int kb = (lane >> 4) * 8;
  v16h b;
#pragma unroll
  for (int e = 0; e < 16; ++e) b[e] = (_Float16)((e < 8) ? k : rk)[(kb + (e & 7)) * 48 + g * 16 + n];
  return b;
}

// Half-zero B operand for the h-gate: upper=false -> only K=0..15 (k) valid,
// upper=true -> only K=16..31 (rk) valid.
__device__ __forceinline__ v16h load_b_half(const float* __restrict__ w,
                                            int g, int lane, bool upper) {
  const int n = lane & 15;
  const int kb = (lane >> 4) * 8;
  v16h b;
#pragma unroll
  for (int e = 0; e < 16; ++e)
    b[e] = (upper == (e >= 8)) ? (_Float16)w[(kb + (e & 7)) * 48 + g * 16 + n] : (_Float16)0.0f;
  return b;
}

// ---------------- link load (utilization) -----------------------------------
__global__ __launch_bounds__(128) void link_load_kernel(
    const float* __restrict__ traffic, const int* __restrict__ ptl,
    const float* __restrict__ cap, float* __restrict__ loadv) {
  __shared__ float red[128];
  const int l = blockIdx.x, tid = threadIdx.x;
  float v = 0.0f;
  if (tid < MAX_PL) v = traffic[ptl[(l * MAX_PL + tid) * 2]];
  red[tid] = v;
  __syncthreads();
  for (int s = 64; s > 0; s >>= 1) {
    if (tid < s) red[tid] += red[tid + s];
    __syncthreads();
  }
  if (tid == 0) loadv[l] = red[0] * fast_rcp(cap[l]) * 1.0e-9f;
}

// ---------------- path encoder ----------------------------------------------
__global__ __launch_bounds__(256) void path_enc_kernel(
    const float* __restrict__ tr, const float* __restrict__ pk,
    const float* __restrict__ psz, const float* __restrict__ ft,
    const float* __restrict__ w1, const float* __restrict__ b1,
    const float* __restrict__ w2, const float* __restrict__ b2,
    float* __restrict__ path_state) {
  __shared__ float W1[80], B1v[16], W2[256], B2v[16];
  const int tid = threadIdx.x;
  if (tid < 80) W1[tid] = w1[tid];
  if (tid < 16) { B1v[tid] = b1[tid]; B2v[tid] = b2[tid]; }
  W2[tid] = w2[tid];
  __syncthreads();
  const int p = blockIdx.x * 256 + tid;
  if (p >= N_PATHS) return;
  float f[5] = { tr[p] * 1.0e-4f, pk[p] * 1.0e-2f, psz[p] * 1.0e-3f, ft[2 * p], ft[2 * p + 1] };
  float h1[16];
#pragma unroll
  for (int j = 0; j < 16; ++j) {
    float s = B1v[j];
#pragma unroll
    for (int i = 0; i < 5; ++i) s += f[i] * W1[i * 16 + j];
    h1[j] = fmaxf(s, 0.0f);
  }
#pragma unroll
  for (int j = 0; j < 16; ++j) {
    float s = B2v[j];
#pragma unroll
    for (int i = 0; i < 16; ++i) s += h1[i] * W2[i * 16 + j];
    path_state[p * 16 + j] = fmaxf(s, 0.0f);
  }
}

// ---------------- link encoder ----------------------------------------------
__global__ __launch_bounds__(256) void link_enc_kernel(
    const float* __restrict__ cap, const float* __restrict__ loadv,
    const float* __restrict__ w1, const float* __restrict__ b1,
    const float* __restrict__ w2, const float* __restrict__ b2,
    float* __restrict__ link_state, _Float16* __restrict__ lf16) {
  __shared__ float W1[32], B1v[16], W2[256], B2v[16];
  const int tid = threadIdx.x;
  if (tid < 32) W1[tid] = w1[tid];
  if (tid < 16) { B1v[tid] = b1[tid]; B2v[tid] = b2[tid]; }
  W2[tid] = w2[tid];
  __syncthreads();
  const int l = blockIdx.x * 256 + tid;
  if (l >= N_LINKS) return;
  const float f0 = cap[l] * 0.1f, f1 = loadv[l];
  float h1[16];
#pragma unroll
  for (int j = 0; j < 16; ++j)
    h1[j] = fmaxf(f0 * W1[j] + f1 * W1[16 + j] + B1v[j], 0.0f);
#pragma unroll
  for (int j = 0; j < 16; ++j) {
    float s = B2v[j];
#pragma unroll
    for (int i = 0; i < 16; ++i) s += h1[i] * W2[i * 16 + j];
    s = fmaxf(s, 0.0f);
    link_state[l * 16 + j] = s;
    lf16[l * 16 + j] = (_Float16)s;
  }
}

// ---------------- path GRU (WMMA, 8 steps per iteration) --------------------
// One wave per 16-path tile; 8 waves per block. A operand = [x | h] (dense
// K=32); fused B tiles compute x@k + h@rk for the z/r gates in one WMMA each;
// the h-gate keeps its two halves separate (r scales only the recurrent part).
// LDS tile (16 rows x 20 dwords, padded so float4 readbacks stay 16B-aligned)
// does the D->A layout transpose. Link indices for all 8 steps are loaded up
// front; step t+1's f16 link state is prefetched during step t's math.
__global__ __launch_bounds__(256) void path_gru_kernel(
    const int* __restrict__ ltp, const _Float16* __restrict__ lf16,
    float* __restrict__ path_state, float* __restrict__ seq,
    const float* __restrict__ gk, const float* __restrict__ grk,
    const float* __restrict__ gb) {
  __shared__ float tile[8][16 * 20];
  const int lane = threadIdx.x & 31;
  const int wv = threadIdx.x >> 5;
  const int tileId = blockIdx.x * 8 + wv;
  if (tileId >= N_PATHS / 16) return;
  const int col = lane & 15;        // A row / D column / bias dim
  const int hi = lane >> 4;
  const int kb = hi * 8;            // dim base this half-wave covers in A layout
  const int prow = tileId * 16 + col;
  float* T = tile[wv];

  const v16h Bzt = load_b_fused(gk, grk, 0, lane);
  const v16h Brt = load_b_fused(gk, grk, 1, lane);
  const v16h Bxh = load_b_half(gk, 2, lane, false);
  const v16h Brh = load_b_half(grk, 2, lane, true);
  const float bzv = gb[col] + gb[48 + col];
  const float brv = gb[16 + col] + gb[64 + col];
  const float b0h = gb[32 + col];
  const float b1h = gb[80 + col];

  // all 8 link indices for this path row (32B contiguous)
  const int4 liA = *(const int4*)(ltp + prow * PATH_LEN);
  const int4 liB = *(const int4*)(ltp + prow * PATH_LEN + 4);
  const int lis[8] = { liA.x, liA.y, liA.z, liA.w, liB.x, liB.y, liB.z, liB.w };

  // h0 = prev path_state ; write seq[:,0,:] ; get D-layout copy via LDS
  float4 lo = *(const float4*)(path_state + prow * 16 + kb);
  float4 hf = *(const float4*)(path_state + prow * 16 + kb + 4);
  *(float4*)(seq + prow * 144 + kb) = lo;
  *(float4*)(seq + prow * 144 + kb + 4) = hf;
  T[col * 20 + kb + 0] = lo.x; T[col * 20 + kb + 1] = lo.y;
  T[col * 20 + kb + 2] = lo.z; T[col * 20 + kb + 3] = lo.w;
  T[col * 20 + kb + 4] = hf.x; T[col * 20 + kb + 5] = hf.y;
  T[col * 20 + kb + 6] = hf.z; T[col * 20 + kb + 7] = hf.w;
  asm volatile("s_wait_dscnt 0" ::: "memory");
  float hD[8];
#pragma unroll
  for (int r = 0; r < 8; ++r) hD[r] = T[(r + 8 * hi) * 20 + col];

  v16h af;               // [x | h] fused A operand; elements 8..15 = h
  af[8]  = (_Float16)lo.x; af[9]  = (_Float16)lo.y; af[10] = (_Float16)lo.z; af[11] = (_Float16)lo.w;
  af[12] = (_Float16)hf.x; af[13] = (_Float16)hf.y; af[14] = (_Float16)hf.z; af[15] = (_Float16)hf.w;

  // prefetch x_0
  v8h xh = *(const v8h*)(lf16 + lis[0] * 16 + kb);

#pragma unroll
  for (int t = 0; t < PATH_LEN; ++t) {
#pragma unroll
    for (int i = 0; i < 8; ++i) af[i] = xh[i];
    if (t + 1 < PATH_LEN)                       // prefetch next step's x
      xh = *(const v8h*)(lf16 + lis[t + 1] * 16 + kb);

    v8f z8 = {};
    v8f cz  = wmma16(af, Bzt, z8);   // x@k_z + h@rk_z
    v8f cr  = wmma16(af, Brt, z8);   // x@k_r + h@rk_r
    v8f cxh = wmma16(af, Bxh, z8);   // x@k_h
    v8f cih = wmma16(af, Brh, z8);   // h@rk_h

#pragma unroll
    for (int r = 0; r < 8; ++r) {
      const float zg = sigmoidf_(cz[r] + bzv);
      const float rg = sigmoidf_(cr[r] + brv);
      const float hh = tanhf_(cxh[r] + b0h + rg * (cih[r] + b1h));
      const float hn = zg * hD[r] + (1.0f - zg) * hh;
      hD[r] = hn;
      T[(r + 8 * hi) * 20 + col] = hn;   // D layout -> LDS
    }
    asm volatile("s_wait_dscnt 0" ::: "memory");
    lo = *(const float4*)&T[col * 20 + kb];        // A-layout readback
    hf = *(const float4*)&T[col * 20 + kb + 4];
    *(float4*)(seq + prow * 144 + (t + 1) * 16 + kb) = lo;
    *(float4*)(seq + prow * 144 + (t + 1) * 16 + kb + 4) = hf;
    af[8]  = (_Float16)lo.x; af[9]  = (_Float16)lo.y; af[10] = (_Float16)lo.z; af[11] = (_Float16)lo.w;
    af[12] = (_Float16)hf.x; af[13] = (_Float16)hf.y; af[14] = (_Float16)hf.z; af[15] = (_Float16)hf.w;
  }
  *(float4*)(path_state + prow * 16 + kb) = lo;
  *(float4*)(path_state + prow * 16 + kb + 4) = hf;
}

// ---------------- per-link gather-sum over up to 100 path states ------------
__global__ __launch_bounds__(128) void path_sum_kernel(
    const int* __restrict__ ptl, const float* __restrict__ seq,
    float* __restrict__ psum) {
  __shared__ float red[128];
  const int l = blockIdx.x, tid = threadIdx.x;
  const int d = tid & 15, sl = tid >> 4;
  float acc = 0.0f;
  for (int j = sl; j < MAX_PL; j += 8) {
    const int p = ptl[(l * MAX_PL + j) * 2];
    const int s = ptl[(l * MAX_PL + j) * 2 + 1];
    acc += seq[(p * 9 + s) * 16 + d];
  }
  red[tid] = acc;
  __syncthreads();
  if (tid < 16) {
    float s2 = 0.0f;
#pragma unroll
    for (int k = 0; k < 8; ++k) s2 += red[tid + 16 * k];
    psum[l * 16 + tid] = s2;
  }
}

// ---------------- link GRU (WMMA, one step) ---------------------------------
__global__ __launch_bounds__(128) void link_gru_kernel(
    const float* __restrict__ psum, float* __restrict__ link_state,
    _Float16* __restrict__ lf16,
    const float* __restrict__ gk, const float* __restrict__ grk,
    const float* __restrict__ gb) {
  __shared__ float tile[4][16 * 20];
  const int lane = threadIdx.x & 31;
  const int wv = threadIdx.x >> 5;
  const int tileId = blockIdx.x * 4 + wv;
  if (tileId >= N_LINKS / 16) return;
  const int col = lane & 15, hi = lane >> 4, kb = hi * 8;
  const int lrow = tileId * 16 + col;
  float* T = tile[wv];

  const v16h Bzt = load_b_fused(gk, grk, 0, lane);
  const v16h Brt = load_b_fused(gk, grk, 1, lane);
  const v16h Bxh = load_b_half(gk, 2, lane, false);
  const v16h Brh = load_b_half(grk, 2, lane, true);
  const float bzv = gb[col] + gb[48 + col];
  const float brv = gb[16 + col] + gb[64 + col];
  const float b0h = gb[32 + col];
  const float b1h = gb[80 + col];

  // A operand = [x | h]: x = path_sum, h = link_state
  float4 xlo = *(const float4*)(psum + lrow * 16 + kb);
  float4 xhf = *(const float4*)(psum + lrow * 16 + kb + 4);
  float4 lo = *(const float4*)(link_state + lrow * 16 + kb);
  float4 hf = *(const float4*)(link_state + lrow * 16 + kb + 4);
  v16h af;
  af[0]  = (_Float16)xlo.x; af[1]  = (_Float16)xlo.y; af[2]  = (_Float16)xlo.z; af[3]  = (_Float16)xlo.w;
  af[4]  = (_Float16)xhf.x; af[5]  = (_Float16)xhf.y; af[6]  = (_Float16)xhf.z; af[7]  = (_Float16)xhf.w;
  af[8]  = (_Float16)lo.x;  af[9]  = (_Float16)lo.y;  af[10] = (_Float16)lo.z;  af[11] = (_Float16)lo.w;
  af[12] = (_Float16)hf.x;  af[13] = (_Float16)hf.y;  af[14] = (_Float16)hf.z;  af[15] = (_Float16)hf.w;

  // D-layout copy of h via LDS
  T[col * 20 + kb + 0] = lo.x; T[col * 20 + kb + 1] = lo.y;
  T[col * 20 + kb + 2] = lo.z; T[col * 20 + kb + 3] = lo.w;
  T[col * 20 + kb + 4] = hf.x; T[col * 20 + kb + 5] = hf.y;
  T[col * 20 + kb + 6] = hf.z; T[col * 20 + kb + 7] = hf.w;
  asm volatile("s_wait_dscnt 0" ::: "memory");
  float hD[8];
#pragma unroll
  for (int r = 0; r < 8; ++r) hD[r] = T[(r + 8 * hi) * 20 + col];

  v8f z8 = {};
  v8f cz  = wmma16(af, Bzt, z8);
  v8f cr  = wmma16(af, Brt, z8);
  v8f cxh = wmma16(af, Bxh, z8);
  v8f cih = wmma16(af, Brh, z8);

#pragma unroll
  for (int r = 0; r < 8; ++r) {
    const float zg = sigmoidf_(cz[r] + bzv);
    const float rg = sigmoidf_(cr[r] + brv);
    const float hh = tanhf_(cxh[r] + b0h + rg * (cih[r] + b1h));
    T[(r + 8 * hi) * 20 + col] = zg * hD[r] + (1.0f - zg) * hh;
  }
  asm volatile("s_wait_dscnt 0" ::: "memory");
  lo = *(const float4*)&T[col * 20 + kb];
  hf = *(const float4*)&T[col * 20 + kb + 4];
  *(float4*)(link_state + lrow * 16 + kb) = lo;
  *(float4*)(link_state + lrow * 16 + kb + 4) = hf;
  v8h hv;
  hv[0] = (_Float16)lo.x; hv[1] = (_Float16)lo.y; hv[2] = (_Float16)lo.z; hv[3] = (_Float16)lo.w;
  hv[4] = (_Float16)hf.x; hv[5] = (_Float16)hf.y; hv[6] = (_Float16)hf.z; hv[7] = (_Float16)hf.w;
  *(v8h*)(lf16 + lrow * 16 + kb) = hv;
}

// ---------------- readout ----------------------------------------------------
__global__ __launch_bounds__(256) void readout_kernel(
    const float* __restrict__ seq, const int* __restrict__ ltp,
    const float* __restrict__ cap,
    const float* __restrict__ w1, const float* __restrict__ b1,
    const float* __restrict__ w2, const float* __restrict__ b2,
    const float* __restrict__ w3, const float* __restrict__ b3,
    float* __restrict__ out) {
  __shared__ float W1[128], B1v[8], W2[32], B2v[4], W3[4], B3;
  const int tid = threadIdx.x;
  if (tid < 128) W1[tid] = w1[tid];
  if (tid < 8) B1v[tid] = b1[tid];
  if (tid < 32) W2[tid] = w2[tid];
  if (tid < 4) { B2v[tid] = b2[tid]; W3[tid] = w3[tid]; }
  if (tid == 0) B3 = b3[0];
  __syncthreads();
  const int p = blockIdx.x * 256 + tid;
  if (p >= N_PATHS) return;
  float acc = 0.0f;
  for (int t = 1; t <= PATH_LEN; ++t) {
    const float* x = seq + p * 144 + t * 16;
    float h1[8];
#pragma unroll
    for (int j = 0; j < 8; ++j) {
      float s = B1v[j];
#pragma unroll
      for (int i = 0; i < 16; ++i) s += x[i] * W1[i * 8 + j];
      h1[j] = fmaxf(s, 0.0f);
    }
    float h2[4];
#pragma unroll
    for (int j = 0; j < 4; ++j) {
      float s = B2v[j];
#pragma unroll
      for (int i = 0; i < 8; ++i) s += h1[i] * W2[i * 4 + j];
      h2[j] = fmaxf(s, 0.0f);
    }
    float o = B3;
#pragma unroll
    for (int i = 0; i < 4; ++i) o += h2[i] * W3[i];
    acc += softplusf_(o) * fast_rcp(cap[ltp[p * PATH_LEN + (t - 1)]]);
  }
  out[p] = acc;
}

extern "C" void kernel_launch(void* const* d_in, const int* in_sizes, int n_in,
                              void* d_out, int out_size, void* d_ws, size_t ws_size,
                              hipStream_t stream) {
  const float* flow_traffic = (const float*)d_in[0];
  const float* flow_packets = (const float*)d_in[1];
  const float* flow_pktsize = (const float*)d_in[2];
  const float* flow_type    = (const float*)d_in[3];
  const float* link_cap     = (const float*)d_in[4];
  const int*   ltp          = (const int*)d_in[5];
  const int*   ptl          = (const int*)d_in[6];
  const float* fe_w1 = (const float*)d_in[7];
  const float* fe_b1 = (const float*)d_in[8];
  const float* fe_w2 = (const float*)d_in[9];
  const float* fe_b2 = (const float*)d_in[10];
  const float* le_w1 = (const float*)d_in[11];
  const float* le_b1 = (const float*)d_in[12];
  const float* le_w2 = (const float*)d_in[13];
  const float* le_b2 = (const float*)d_in[14];
  const float* pgru_k  = (const float*)d_in[15];
  const float* pgru_rk = (const float*)d_in[16];
  const float* pgru_b  = (const float*)d_in[17];
  const float* lgru_k  = (const float*)d_in[18];
  const float* lgru_rk = (const float*)d_in[19];
  const float* lgru_b  = (const float*)d_in[20];
  const float* ro_w1 = (const float*)d_in[21];
  const float* ro_b1 = (const float*)d_in[22];
  const float* ro_w2 = (const float*)d_in[23];
  const float* ro_b2 = (const float*)d_in[24];
  const float* ro_w3 = (const float*)d_in[25];
  const float* ro_b3 = (const float*)d_in[26];

  char* w = (char*)d_ws;
  auto carve = [&](size_t bytes) { void* p = (void*)w; w += (bytes + 255) & ~(size_t)255; return p; };
  float*     path_state = (float*)carve((size_t)N_PATHS * 16 * 4);
  float*     seq        = (float*)carve((size_t)N_PATHS * 9 * 16 * 4);
  float*     link_state = (float*)carve((size_t)N_LINKS * 16 * 4);
  _Float16*  lf16       = (_Float16*)carve((size_t)N_LINKS * 16 * 2);
  float*     psum       = (float*)carve((size_t)N_LINKS * 16 * 4);
  float*     loadv      = (float*)carve((size_t)N_LINKS * 4);

  link_load_kernel<<<N_LINKS, 128, 0, stream>>>(flow_traffic, ptl, link_cap, loadv);
  path_enc_kernel<<<(N_PATHS + 255) / 256, 256, 0, stream>>>(
      flow_traffic, flow_packets, flow_pktsize, flow_type,
      fe_w1, fe_b1, fe_w2, fe_b2, path_state);
  link_enc_kernel<<<(N_LINKS + 255) / 256, 256, 0, stream>>>(
      link_cap, loadv, le_w1, le_b1, le_w2, le_b2, link_state, lf16);

  const int path_tiles = N_PATHS / 16;   // 12500
  const int link_tiles = N_LINKS / 16;   // 1250
  for (int it = 0; it < ITERS; ++it) {
    path_gru_kernel<<<(path_tiles + 7) / 8, 256, 0, stream>>>(
        ltp, lf16, path_state, seq, pgru_k, pgru_rk, pgru_b);
    path_sum_kernel<<<N_LINKS, 128, 0, stream>>>(ptl, seq, psum);
    link_gru_kernel<<<(link_tiles + 3) / 4, 128, 0, stream>>>(
        psum, link_state, lf16, lgru_k, lgru_rk, lgru_b);
  }
  readout_kernel<<<(N_PATHS + 255) / 256, 256, 0, stream>>>(
      seq, ltp, link_cap, ro_w1, ro_b1, ro_w2, ro_b2, ro_w3, ro_b3, (float*)d_out);
}
